// HierarchicalAttention_34454227648862
// MI455X (gfx1250) — matrix operand order repaired and checked by reading, not compiled, not run on hardware
//
#include <hip/hip_runtime.h>
#include <hip/hip_bf16.h>

typedef __attribute__((ext_vector_type(16))) _Float16 v16h;
typedef __attribute__((ext_vector_type(8)))  _Float16 v8h;
typedef __attribute__((ext_vector_type(8)))  float    v8f;

#define BATCH 8
#define SLEN  1000
#define HID   768
#define NHEAD 12
#define HD    64
#define SVP   1024          // padded V stride in s
#define QTILES 63           // ceil(1000/16)
#define MTILES 500          // 8000/16
#define NTILES 12           // 768/64

__device__ __forceinline__ v8f wmma16(v16h a, v16h b, v8f c) {
  return __builtin_amdgcn_wmma_f32_16x16x32_f16(false, a, false, b, (short)0, c, false, false);
}
__device__ __forceinline__ v16h mk16(v8h lo, v8h hi) {
  v16h r;
#pragma unroll
  for (int i = 0; i < 8; ++i) { r[i] = lo[i]; r[i + 8] = hi[i]; }
  return r;
}
__device__ __forceinline__ void wave_lds_fence() {
  __builtin_amdgcn_wave_barrier();
  asm volatile("s_wait_dscnt 0x0" ::: "memory");
  __builtin_amdgcn_wave_barrier();
}
__device__ __forceinline__ int iabs_i(int x) { return x < 0 ? -x : x; }

// ---------------- elementwise helpers ----------------
__global__ void cvt_f32_to_f16_kernel(const float* __restrict__ src,
                                      _Float16* __restrict__ dst, int n) {
  int i = blockIdx.x * blockDim.x + threadIdx.x;
  if (i < n) dst[i] = (_Float16)src[i];
}

__global__ void zero_f16_kernel(_Float16* __restrict__ dst, int n) {
  int i = blockIdx.x * blockDim.x + threadIdx.x;
  if (i < n) dst[i] = (_Float16)0.0f;
}

// x = features + pos_emb * (1 + 0.1*level), stored f16 [8000][768]
__global__ void embed_kernel(const float* __restrict__ feat,
                             const float* __restrict__ pos,
                             const int* __restrict__ level,
                             _Float16* __restrict__ x, int n) {
  int i = blockIdx.x * blockDim.x + threadIdx.x;
  if (i >= n) return;
  float scale = 1.0f + 0.1f * (float)level[0];
  int sh = i % (SLEN * HID);
  x[i] = (_Float16)(feat[i] + pos[sh] * scale);
}

// ---------------- WMMA GEMM: Y[8000,768] = X[8000,768] @ W[768,768]^T + b ----
// mode 0: out f16 [b, head, s, 64]          (Q, K)
// mode 1: out f16 [b, head, 64, SVP] (V^T, s-contiguous, padded)
// mode 2: out f32 [8000,768] = acc + bias + resid (output projection)
__global__ __launch_bounds__(128)
void gemm_kernel(const _Float16* __restrict__ A, const _Float16* __restrict__ W,
                 const float* __restrict__ bias, const float* __restrict__ resid,
                 void* __restrict__ outp, int mode) {
  const int lane = threadIdx.x & 31;
  const int wv   = threadIdx.x >> 5;
  const int tile = blockIdx.x * 4 + wv;          // 6000 tiles exactly
  const int mt = tile % MTILES, nt = tile / MTILES;
  const int m0 = mt * 16, n0 = nt * 64;
  const int r = lane & 15, g = lane >> 4;
  const int kb8 = g * 8, kb16 = g * 16;

  const _Float16* arow = A + (size_t)(m0 + r) * HID;
  const _Float16* wr0 = W + (size_t)(n0 +  0 + r) * HID + kb16;
  const _Float16* wr1 = W + (size_t)(n0 + 16 + r) * HID + kb16;
  const _Float16* wr2 = W + (size_t)(n0 + 32 + r) * HID + kb16;
  const _Float16* wr3 = W + (size_t)(n0 + 48 + r) * HID + kb16;

  v8f acc0 = {}, acc1 = {}, acc2 = {}, acc3 = {};

  // software pipeline: double-buffered fragments so iteration k+1's loads
  // are in flight while iteration k's WMMAs execute.
  v16h ac = mk16(*(const v8h*)(arow + kb16), *(const v8h*)(arow + kb16 + 8));
  v16h b0c = *(const v16h*)(wr0);
  v16h b1c = *(const v16h*)(wr1);
  v16h b2c = *(const v16h*)(wr2);
  v16h b3c = *(const v16h*)(wr3);

  for (int k0 = 0; k0 < HID; k0 += 32) {
    const int kn = (k0 + 32 < HID) ? (k0 + 32) : 0;   // wrap: dead loads on last iter
    v16h an  = mk16(*(const v8h*)(arow + kn + kb16),
                    *(const v8h*)(arow + kn + kb16 + 8));
    v16h b0n = *(const v16h*)(wr0 + kn);
    v16h b1n = *(const v16h*)(wr1 + kn);
    v16h b2n = *(const v16h*)(wr2 + kn);
    v16h b3n = *(const v16h*)(wr3 + kn);
    __builtin_prefetch((const void*)(arow + kn + 64), 0, 1);

    acc0 = wmma16(ac, b0c, acc0);
    acc1 = wmma16(ac, b1c, acc1);
    acc2 = wmma16(ac, b2c, acc2);
    acc3 = wmma16(ac, b3c, acc3);

    ac = an; b0c = b0n; b1c = b1n; b2c = b2n; b3c = b3n;
  }

  float bv4[4];
#pragma unroll
  for (int c = 0; c < 4; ++c) bv4[c] = bias[n0 + c * 16 + r];

#pragma unroll
  for (int v = 0; v < 8; ++v) {
    int mrow = m0 + v + 8 * g;                   // < 8000 always
    float vals[4] = { acc0[v] + bv4[0], acc1[v] + bv4[1],
                      acc2[v] + bv4[2], acc3[v] + bv4[3] };
    if (mode == 2) {
      float* Y = (float*)outp;
      const float* rr = resid + (size_t)mrow * HID + n0;
#pragma unroll
      for (int c = 0; c < 4; ++c)
        Y[(size_t)mrow * HID + n0 + c * 16 + r] = vals[c] + rr[c * 16 + r];
    } else {
      _Float16* O = (_Float16*)outp;
      int bb = mrow / SLEN, ss = mrow % SLEN;
      if (mode == 0) {
        _Float16* orow = O + (((size_t)bb * NHEAD + nt) * SLEN + ss) * HD;
#pragma unroll
        for (int c = 0; c < 4; ++c) orow[c * 16 + r] = (_Float16)vals[c];
      } else {
        _Float16* ob = O + ((size_t)bb * NHEAD + nt) * HD * SVP + ss;
#pragma unroll
        for (int c = 0; c < 4; ++c)
          ob[(size_t)(c * 16 + r) * SVP] = (_Float16)vals[c];
      }
    }
  }
}

// ---------------- banded flash attention ----------------
// one wave per (b, head, 16-query tile); streams 32-wide key tiles in band
__global__ __launch_bounds__(128)
void attn_kernel(const _Float16* __restrict__ Q, const _Float16* __restrict__ K,
                 const _Float16* __restrict__ Vt, const int* __restrict__ level_p,
                 _Float16* __restrict__ ctx) {
  __shared__ _Float16 lds_p[4][16][32];
  const int lane = threadIdx.x & 31;
  const int wv   = threadIdx.x >> 5;
  const int tile = blockIdx.x * 4 + wv;          // 6048 tiles exactly
  const int qt = tile % QTILES;
  const int hh = (tile / QTILES) % NHEAD;
  const int bb = tile / (QTILES * NHEAD);
  const int q0 = qt * 16;
  const int lev = level_p[0];
  const int w = (lev == 0) ? 2 : ((lev == 1) ? 5 : SLEN);

  const _Float16* Qh = Q  + ((size_t)bb * NHEAD + hh) * SLEN * HD;
  const _Float16* Kh = K  + ((size_t)bb * NHEAD + hh) * SLEN * HD;
  const _Float16* Vh = Vt + ((size_t)bb * NHEAD + hh) * HD * SVP;

  const int r = lane & 15, g = lane >> 4;
  const int kb8 = g * 8, kb16 = g * 16;

  int qi = q0 + r; if (qi > SLEN - 1) qi = SLEN - 1;
  const _Float16* qrow = Qh + (size_t)qi * HD;
  v16h qa0 = mk16(*(const v8h*)(qrow + kb16),      *(const v8h*)(qrow + kb16 + 8));
  v16h qa1 = mk16(*(const v8h*)(qrow + 32 + kb16), *(const v8h*)(qrow + 32 + kb16 + 8));

  v8f acc0 = {}, acc1 = {}, acc2 = {}, acc3 = {};
  float m_run[8], l_run[8];
#pragma unroll
  for (int v = 0; v < 8; ++v) { m_run[v] = -INFINITY; l_run[v] = 0.0f; }
  const float sm = 0.125f;                        // 1/sqrt(64)

  int jlo = q0 - w; if (jlo < 0) jlo = 0; jlo &= ~31;
  int jhi = q0 + 15 + w; if (jhi > SLEN - 1) jhi = SLEN - 1;

  for (int j0 = jlo; j0 <= jhi; j0 += 32) {
    int k0i = j0 + r;      if (k0i > SLEN - 1) k0i = SLEN - 1;
    int k1i = j0 + 16 + r; if (k1i > SLEN - 1) k1i = SLEN - 1;
    v16h kb0lo = *(const v16h*)(Kh + (size_t)k0i * HD + kb16);
    v16h kb0hi = *(const v16h*)(Kh + (size_t)k0i * HD + 32 + kb16);
    v16h kb1lo = *(const v16h*)(Kh + (size_t)k1i * HD + kb16);
    v16h kb1hi = *(const v16h*)(Kh + (size_t)k1i * HD + 32 + kb16);

    // V fragments issued early: independent of softmax, so these loads
    // overlap the QK^T WMMAs and the entire exp/shuffle VALU section.
    const _Float16* vbase = Vh + j0 + kb16;
    v16h vb0 = *(const v16h*)(vbase + (size_t)( 0 + r) * SVP);
    v16h vb1 = *(const v16h*)(vbase + (size_t)(16 + r) * SVP);
    v16h vb2 = *(const v16h*)(vbase + (size_t)(32 + r) * SVP);
    v16h vb3 = *(const v16h*)(vbase + (size_t)(48 + r) * SVP);

    v8f s0 = {}, s1 = {};
    s0 = wmma16(qa0, kb0lo, s0);  s0 = wmma16(qa1, kb0hi, s0);
    s1 = wmma16(qa0, kb1lo, s1);  s1 = wmma16(qa1, kb1hi, s1);

    float p0[8], p1[8], sc[8];
#pragma unroll
    for (int v = 0; v < 8; ++v) {
      int ic = q0 + v + 8 * g;
      int jc0 = j0 + r, jc1 = j0 + 16 + r;
      float e0 = (jc0 < SLEN && ic < SLEN && iabs_i(ic - jc0) <= w)
                   ? s0[v] * sm : -INFINITY;
      float e1 = (jc1 < SLEN && ic < SLEN && iabs_i(ic - jc1) <= w)
                   ? s1[v] * sm : -INFINITY;
      float t = fmaxf(e0, e1);
#pragma unroll
      for (int msk = 1; msk <= 8; msk <<= 1) t = fmaxf(t, __shfl_xor(t, msk, 32));
      float mold = m_run[v];
      float mnew = fmaxf(mold, t);
      float mc   = fmaxf(mnew, -1e30f);
      float scl  = __expf(mold - mc);             // -inf -> 0
      float q0e  = __expf(e0 - mc);
      float q1e  = __expf(e1 - mc);
      float rs   = q0e + q1e;
#pragma unroll
      for (int msk = 1; msk <= 8; msk <<= 1) rs += __shfl_xor(rs, msk, 32);
      l_run[v] = l_run[v] * scl + rs;
      m_run[v] = mnew;
      sc[v] = scl; p0[v] = q0e; p1[v] = q1e;
    }
#pragma unroll
    for (int v = 0; v < 8; ++v) {
      acc0[v] *= sc[v]; acc1[v] *= sc[v]; acc2[v] *= sc[v]; acc3[v] *= sc[v];
    }
    // C-layout -> A-layout transpose of P through wave-private LDS
#pragma unroll
    for (int v = 0; v < 8; ++v) {
      int row = v + 8 * g;
      lds_p[wv][row][r]      = (_Float16)p0[v];
      lds_p[wv][row][r + 16] = (_Float16)p1[v];
    }
    wave_lds_fence();
    const _Float16* prow = &lds_p[wv][r][0];
    v16h pa = mk16(*(const v8h*)(prow + kb16), *(const v8h*)(prow + kb16 + 8));
    wave_lds_fence();

    acc0 = wmma16(pa, vb0, acc0);
    acc1 = wmma16(pa, vb1, acc1);
    acc2 = wmma16(pa, vb2, acc2);
    acc3 = wmma16(pa, vb3, acc3);
  }

#pragma unroll
  for (int v = 0; v < 8; ++v) {
    int ic = q0 + v + 8 * g;
    if (ic >= SLEN) continue;
    float inv = l_run[v] > 0.0f ? 1.0f / l_run[v] : 0.0f;
    _Float16* crow = ctx + ((size_t)bb * SLEN + ic) * HID + hh * HD;
    crow[ 0 + r] = (_Float16)(acc0[v] * inv);
    crow[16 + r] = (_Float16)(acc1[v] * inv);
    crow[32 + r] = (_Float16)(acc2[v] * inv);
    crow[48 + r] = (_Float16)(acc3[v] * inv);
  }
}

// ---------------- LayerNorm over H=768, one block per row ----------------
__global__ __launch_bounds__(256)
void ln_kernel(const float* __restrict__ Y, const float* __restrict__ G,
               const float* __restrict__ Bv, float* __restrict__ out) {
  __shared__ float s_sum[256], s_sq[256];
  const int row = blockIdx.x, tid = threadIdx.x;
  const float* yr = Y + (size_t)row * HID;
  float vals[3], a0 = 0.0f, a1 = 0.0f;
#pragma unroll
  for (int t = 0; t < 3; ++t) {
    float x = yr[tid + 256 * t];
    vals[t] = x; a0 += x; a1 += x * x;
  }
  s_sum[tid] = a0; s_sq[tid] = a1;
  __syncthreads();
  for (int st = 128; st > 0; st >>= 1) {
    if (tid < st) { s_sum[tid] += s_sum[tid + st]; s_sq[tid] += s_sq[tid + st]; }
    __syncthreads();
  }
  float mu   = s_sum[0] * (1.0f / HID);
  float var  = s_sq[0] * (1.0f / HID) - mu * mu;
  float rstd = rsqrtf(var + 1e-5f);
#pragma unroll
  for (int t = 0; t < 3; ++t) {
    int col = tid + 256 * t;
    out[(size_t)row * HID + col] = (vals[t] - mu) * rstd * G[col] + Bv[col];
  }
}

// ---------------- host-side orchestration ----------------
extern "C" void kernel_launch(void* const* d_in, const int* in_sizes, int n_in,
                              void* d_out, int out_size, void* d_ws, size_t ws_size,
                              hipStream_t stream) {
  const float* feat = (const float*)d_in[0];
  const float* pos  = (const float*)d_in[1];
  const float* Wq   = (const float*)d_in[2];
  const float* bq   = (const float*)d_in[3];
  const float* Wk   = (const float*)d_in[4];
  const float* bk   = (const float*)d_in[5];
  const float* Wv   = (const float*)d_in[6];
  const float* bv   = (const float*)d_in[7];
  const float* Wo   = (const float*)d_in[8];
  const float* bo   = (const float*)d_in[9];
  const float* ln_g = (const float*)d_in[10];
  const float* ln_b = (const float*)d_in[11];
  const int*   lvl  = (const int*)d_in[12];
  float* out = (float*)d_out;

  char* ws = (char*)d_ws;
  const size_t SZ_ACT  = (size_t)BATCH * SLEN * HID * 2;     // 12,288,000 B
  const size_t SZ_W    = (size_t)HID * HID * 2;              //  1,179,648 B
  const size_t SZ_VT   = (size_t)BATCH * NHEAD * HD * SVP * 2;
  _Float16* xf   = (_Float16*)(ws);
  _Float16* wqf  = (_Float16*)(ws + SZ_ACT);
  _Float16* wkf  = (_Float16*)(ws + SZ_ACT + SZ_W);
  _Float16* wvf  = (_Float16*)(ws + SZ_ACT + 2 * SZ_W);
  _Float16* wof  = (_Float16*)(ws + SZ_ACT + 3 * SZ_W);
  _Float16* Qf   = (_Float16*)(ws + SZ_ACT + 4 * SZ_W);
  _Float16* Kf   = (_Float16*)(ws + 2 * SZ_ACT + 4 * SZ_W);
  _Float16* Vtf  = (_Float16*)(ws + 3 * SZ_ACT + 4 * SZ_W);
  _Float16* ctxf = (_Float16*)(ws + 3 * SZ_ACT + 4 * SZ_W + SZ_VT);
  float*    yf   = (float*)   (ws + 4 * SZ_ACT + 4 * SZ_W + SZ_VT);

  const int nW = HID * HID;                       // 589824
  const int nX = BATCH * SLEN * HID;              // 6144000
  const int nV = BATCH * NHEAD * HD * SVP;        // 6291456

  cvt_f32_to_f16_kernel<<<(nW + 255) / 256, 256, 0, stream>>>(Wq, wqf, nW);
  cvt_f32_to_f16_kernel<<<(nW + 255) / 256, 256, 0, stream>>>(Wk, wkf, nW);
  cvt_f32_to_f16_kernel<<<(nW + 255) / 256, 256, 0, stream>>>(Wv, wvf, nW);
  cvt_f32_to_f16_kernel<<<(nW + 255) / 256, 256, 0, stream>>>(Wo, wof, nW);
  embed_kernel<<<(nX + 255) / 256, 256, 0, stream>>>(feat, pos, lvl, xf, nX);
  zero_f16_kernel<<<(nV + 255) / 256, 256, 0, stream>>>(Vtf, nV);

  const int gemmBlocks = (MTILES * NTILES) / 4;   // 1500
  gemm_kernel<<<gemmBlocks, 128, 0, stream>>>(xf, wqf, bq, nullptr, Qf,  0);
  gemm_kernel<<<gemmBlocks, 128, 0, stream>>>(xf, wkf, bk, nullptr, Kf,  0);
  gemm_kernel<<<gemmBlocks, 128, 0, stream>>>(xf, wvf, bv, nullptr, Vtf, 1);

  const int attnBlocks = (BATCH * NHEAD * QTILES) / 4;  // 1512
  attn_kernel<<<attnBlocks, 128, 0, stream>>>(Qf, Kf, Vtf, lvl, ctxf);

  gemm_kernel<<<gemmBlocks, 128, 0, stream>>>(ctxf, wof, bo, feat, yf, 2);

  ln_kernel<<<BATCH * SLEN, 256, 0, stream>>>(yf, ln_g, ln_b, out);
}
